// LayoutAttention_64647847739404
// MI455X (gfx1250) — matrix operand order repaired and checked by reading, not compiled, not run on hardware
//
#include <hip/hip_runtime.h>
#include <hip/hip_bf16.h>

// ---------------------------------------------------------------------------
// LayoutAttention for MI455X (gfx1250): flash-attention with f16 WMMA,
// double-buffered async (memory -> LDS) K/V staging when available.
// ---------------------------------------------------------------------------

typedef _Float16 half_t;
typedef __attribute__((ext_vector_type(16))) _Float16 v16h;
typedef __attribute__((ext_vector_type(8)))  _Float16 v8h;
typedef __attribute__((ext_vector_type(8)))  float    v8f;
typedef __attribute__((ext_vector_type(4)))  int      v4i;

#define N_TOK  4096
#define QDIM   320
#define INNER  512
#define NHEADS 8
#define DH     64
#define PHASES 3

#if defined(__has_builtin)
#if __has_builtin(__builtin_amdgcn_global_load_async_to_lds_b128)
#define HAS_ASYNC_LDS 1
#else
#define HAS_ASYNC_LDS 0
#endif
#if __has_builtin(__builtin_amdgcn_s_wait_asynccnt)
#define HAS_WAIT_ASYNC 1
#else
#define HAS_WAIT_ASYNC 0
#endif
#else
#define HAS_ASYNC_LDS 0
#define HAS_WAIT_ASYNC 0
#endif

// ---- WMMA wrapper --------------------------------------------------------
__device__ __forceinline__ v8f wmma_f16(v16h a, v16h b, v8f c) {
    return __builtin_amdgcn_wmma_f32_16x16x32_f16(
        /*neg_a=*/false, a, /*neg_b=*/false, b,
        /*c_mod=*/(short)0, c, /*reuse_a=*/false, /*reuse_b=*/false);
}

// A fragment (16x32 f16, row-major source): lane hg=lane/16, lm=lane%16
// lanes 0-15: row=lm, K in {k0+0..7, k0+16..23}; lanes 16-31: K shifted by 8.
__device__ __forceinline__ v16h load_a_frag(const half_t* base, int ld,
                                            int row, int k0, int hg) {
    union { v16h v; v8h h[2]; } u;
    const half_t* p = base + (size_t)row * ld + k0 + 8 * hg;
    u.h[0] = *(const v8h*)(p);
    u.h[1] = *(const v8h*)(p + 16);
    return u.v;
}

// B fragment (32x16 f16) from K-major storage T[n][k] (row-major, ld):
// lane (hg, lm): column n=lm(+tile), K in {k0+8hg+0..7, k0+16+8hg+0..7} -> two 16B loads.
__device__ __forceinline__ v16h load_b_frag(const half_t* rowptr, int k0, int hg) {
    union { v16h v; v8h h[2]; } u;
    const half_t* p = rowptr + k0 + 8 * hg;
    u.h[0] = *(const v8h*)(p);
    u.h[1] = *(const v8h*)(p + 16);
    return u.v;
}

// ---- async 16B global->LDS copy (falls back to sync vector copy) ---------
__device__ __forceinline__ void cp16_g2l(const half_t* g, half_t* l) {
#if HAS_ASYNC_LDS
    // generic pointers: clang inserts addrspacecast to AS1 (src) / AS3 (dst)
    __builtin_amdgcn_global_load_async_to_lds_b128((v4i*)g, (v4i*)l, 0, 0);
#else
    *(v16h*)l = *(const v16h*)g;
#endif
}

__device__ __forceinline__ void wait_async_le8() {
#if HAS_ASYNC_LDS
#if HAS_WAIT_ASYNC
    __builtin_amdgcn_s_wait_asynccnt(8);
#else
    asm volatile("s_wait_asynccnt 0x8" ::: "memory");
#endif
#endif
}

// ---- small prep kernels --------------------------------------------------
__global__ void k_cvt_f16(const float* __restrict__ src, half_t* __restrict__ dst, int n) {
    int i = blockIdx.x * 256 + threadIdx.x;
    if (i < n) dst[i] = (half_t)src[i];
}

// W[K][Nc] f32 -> WT[Nc][K] f16 (K-major for B fragments)
__global__ void k_transpose_f16(const float* __restrict__ W, half_t* __restrict__ WT,
                                int K, int Nc) {
    int i = blockIdx.x * 256 + threadIdx.x;
    if (i < K * Nc) {
        int k = i / Nc, n = i % Nc;
        WT[(size_t)n * K + k] = (half_t)W[i];
    }
}

// guidance_mask [P][HW] (binary f32) -> 3-bit phase code per position
__global__ void k_codes(const float* __restrict__ gm, unsigned int* __restrict__ codes) {
    int i = blockIdx.x * 256 + threadIdx.x;
    if (i < N_TOK) {
        unsigned int c = 0;
#pragma unroll
        for (int p = 0; p < PHASES; ++p)
            if (gm[p * N_TOK + i] > 0.5f) c |= (1u << p);
        codes[i] = c;
    }
}

// ---- QKV projection: xh[4096][320] @ WT -> Q/K (head,n,d) and V^T (head,d,n)
__global__ __launch_bounds__(128) void k_qkv(const half_t* __restrict__ xh,
                                             const half_t* __restrict__ WqT,
                                             const half_t* __restrict__ WkT,
                                             const half_t* __restrict__ WvT,
                                             half_t* __restrict__ Qg,
                                             half_t* __restrict__ Kg,
                                             half_t* __restrict__ VTg) {
    const int mode = blockIdx.z;                 // 0=Q 1=K 2=V
    const half_t* WT = (mode == 0) ? WqT : (mode == 1) ? WkT : WvT;
    const int w    = threadIdx.x >> 5;
    const int lane = threadIdx.x & 31;
    const int hg   = lane >> 4;
    const int lm   = lane & 15;
    const int mt   = blockIdx.x * 16;            // token tile
    const int nt   = blockIdx.y * 64 + w * 16;   // inner-dim tile

    v8f acc = {};
    const int arow = mt + lm;
    const half_t* brow = WT + (size_t)(nt + lm) * QDIM;
#pragma unroll
    for (int kk = 0; kk < QDIM; kk += 32) {
        v16h a = load_a_frag(xh, QDIM, arow, kk, hg);
        v16h b = load_b_frag(brow, kk, hg);
        acc = wmma_f16(a, b, acc);
    }
    const int n = nt + lm;
    const int head = n >> 6, d = n & 63;
#pragma unroll
    for (int r = 0; r < 8; ++r) {
        const int m = mt + r + 8 * hg;
        float v = acc[r];
        if (mode == 0) {
            Qg[((size_t)head * N_TOK + m) * DH + d] = (half_t)(v * 0.125f); // fold 1/sqrt(DH)
        } else if (mode == 1) {
            Kg[((size_t)head * N_TOK + m) * DH + d] = (half_t)v;
        } else {
            VTg[((size_t)head * DH + d) * N_TOK + m] = (half_t)v;          // V transposed
        }
    }
}

// ---- Flash attention: 4 waves/WG, each wave owns 16 query rows -----------
__global__ __launch_bounds__(128) void k_attn(const half_t* __restrict__ Qg,
                                              const half_t* __restrict__ Kg,
                                              const half_t* __restrict__ VTg,
                                              const unsigned int* __restrict__ codes,
                                              half_t* __restrict__ AO) {
    __shared__ half_t Kl[2][64 * DH];    // keys as rows      [j][d], double buffered
    __shared__ half_t Vl[2][DH * 64];    // V transposed tile [d][j], double buffered
    __shared__ half_t Pl[4 * 16 * 32];   // per-wave P relayout scratch

    const int tid  = threadIdx.x;
    const int w    = tid >> 5;
    const int lane = tid & 31;
    const int hg   = lane >> 4;
    const int lm   = lane & 15;
    const int head = blockIdx.y;
    const int qw   = blockIdx.x * 64 + w * 16;   // wave's 16 query rows

    const half_t* Qh = Qg  + (size_t)head * N_TOK * DH;
    const half_t* Kh = Kg  + (size_t)head * N_TOK * DH;
    const half_t* Vh = VTg + (size_t)head * DH * N_TOK;

    // loop-invariant Q fragments (DH = 64 -> two K=32 fragments)
    const v16h qA0 = load_a_frag(Qh, DH, qw + lm, 0,  hg);
    const v16h qA1 = load_a_frag(Qh, DH, qw + lm, 32, hg);

    unsigned int cq[8];
#pragma unroll
    for (int r = 0; r < 8; ++r) cq[r] = codes[qw + r + 8 * hg];

    float mrow[8], lrow[8];
#pragma unroll
    for (int r = 0; r < 8; ++r) { mrow[r] = -1e30f; lrow[r] = 0.0f; }
    v8f o0 = {}, o1 = {}, o2 = {}, o3 = {};

    half_t* pw = Pl + w * (16 * 32);

    // tile stage: 128 threads x 8 x 16B async copies = 16KB (K + V^T)
    const int sj   = tid >> 1;
    const int sseg = (tid & 1) * 32;
    auto stage = [&](int buf, int kt) {
        const half_t* sk = Kh + (size_t)(kt + sj) * DH + sseg;
        const half_t* sv = Vh + (size_t)sj * N_TOK + kt + sseg;
        half_t* dk = Kl[buf] + sj * DH + sseg;
        half_t* dv = Vl[buf] + sj * 64 + sseg;
#pragma unroll
        for (int c = 0; c < 4; ++c) {
            cp16_g2l(sk + c * 8, dk + c * 8);
            cp16_g2l(sv + c * 8, dv + c * 8);
        }
#if !HAS_ASYNC_LDS
        __builtin_prefetch(sk + 64 * DH, 0, 0);
        __builtin_prefetch(sv + 64, 0, 0);
#endif
    };

    const int NT = N_TOK / 64;
    stage(0, 0);                                  // prologue: tile 0 in flight

    for (int it = 0; it < NT; ++it) {
        const int kt  = it * 64;
        const int ktn = (it + 1 < NT) ? kt + 64 : kt;   // clamp (redundant last issue)
        stage((it + 1) & 1, ktn);                 // prefetch next tile into other buffer
        wait_async_le8();                         // current tile's 8 async ops retired
        __syncthreads();

        const half_t* Kc = Kl[it & 1];
        const half_t* Vc = Vl[it & 1];

#pragma unroll
        for (int t = 0; t < 2; ++t) {                // two 32-key chunks
            // ---- S = Q K^T (scale pre-folded into Q) ----
            v8f s0 = {}, s1 = {};
            const half_t* kr0 = Kc + (t * 32 + lm) * DH;
            const half_t* kr1 = Kc + (t * 32 + 16 + lm) * DH;
            s0 = wmma_f16(qA0, load_b_frag(kr0, 0, hg), s0);
            s0 = wmma_f16(qA1, load_b_frag(kr0, 32, hg), s0);
            s1 = wmma_f16(qA0, load_b_frag(kr1, 0, hg), s1);
            s1 = wmma_f16(qA1, load_b_frag(kr1, 32, hg), s1);

            const unsigned int ck0 = codes[kt + t * 32 + lm];
            const unsigned int ck1 = codes[kt + t * 32 + 16 + lm];

            // ---- masked online softmax ----
            float p0[8], p1[8];
#pragma unroll
            for (int r = 0; r < 8; ++r) {
                float a = ((cq[r] & ck0) == 0) ? -1e30f : s0[r];
                float b = ((cq[r] & ck1) == 0) ? -1e30f : s1[r];
                float mx = fmaxf(a, b);
#pragma unroll
                for (int off = 8; off >= 1; off >>= 1)
                    mx = fmaxf(mx, __shfl_xor(mx, off));   // stays within 16-lane half
                const float mnew = fmaxf(mrow[r], mx);
                const float corr = __expf(mrow[r] - mnew);
                const float e0 = __expf(a - mnew);
                const float e1 = __expf(b - mnew);
                float rs = e0 + e1;
#pragma unroll
                for (int off = 8; off >= 1; off >>= 1)
                    rs += __shfl_xor(rs, off);
                lrow[r] = lrow[r] * corr + rs;
                mrow[r] = mnew;
                o0[r] *= corr; o1[r] *= corr; o2[r] *= corr; o3[r] *= corr;
                p0[r] = e0; p1[r] = e1;
            }

            // ---- relayout P (C-layout -> A-layout) through per-wave LDS ----
#pragma unroll
            for (int r = 0; r < 8; ++r) {
                pw[(r + 8 * hg) * 32 + lm]      = (half_t)p0[r];
                pw[(r + 8 * hg) * 32 + 16 + lm] = (half_t)p1[r];
            }
            asm volatile("s_wait_dscnt 0" ::: "memory");   // wave-local LDS RAW
            union { v16h v; v8h h[2]; } up;
            up.h[0] = *(const v8h*)(pw + lm * 32 + 8 * hg);
            up.h[1] = *(const v8h*)(pw + lm * 32 + 8 * hg + 16);
            const v16h pf = up.v;

            // ---- O += P V ----
            o0 = wmma_f16(pf, load_b_frag(Vc + (0 * 16 + lm) * 64, t * 32, hg), o0);
            o1 = wmma_f16(pf, load_b_frag(Vc + (1 * 16 + lm) * 64, t * 32, hg), o1);
            o2 = wmma_f16(pf, load_b_frag(Vc + (2 * 16 + lm) * 64, t * 32, hg), o2);
            o3 = wmma_f16(pf, load_b_frag(Vc + (3 * 16 + lm) * 64, t * 32, hg), o3);
        }
        __syncthreads();   // protect buffer (it&1) before it is re-staged
    }

    // epilogue: normalize, store f16 to AO[n][INNER]
#pragma unroll
    for (int r = 0; r < 8; ++r) {
        const int row = qw + r + 8 * hg;
        const float inv = 1.0f / lrow[r];
        const int cb = head * DH + lm;
        AO[(size_t)row * INNER + cb + 0]  = (half_t)(o0[r] * inv);
        AO[(size_t)row * INNER + cb + 16] = (half_t)(o1[r] * inv);
        AO[(size_t)row * INNER + cb + 32] = (half_t)(o2[r] * inv);
        AO[(size_t)row * INNER + cb + 48] = (half_t)(o3[r] * inv);
    }
}

// ---- output projection: AO[4096][512] @ Wout + bout -> f32 ---------------
__global__ __launch_bounds__(128) void k_proj(const half_t* __restrict__ AO,
                                              const half_t* __restrict__ WoT,
                                              const float* __restrict__ bout,
                                              float* __restrict__ out) {
    const int w    = threadIdx.x >> 5;
    const int lane = threadIdx.x & 31;
    const int hg   = lane >> 4;
    const int lm   = lane & 15;
    const int mt   = blockIdx.x * 16;
    const int nt   = blockIdx.y * 64 + w * 16;   // 320/64 = 5 blocks.y

    v8f acc = {};
    const half_t* brow = WoT + (size_t)(nt + lm) * INNER;
#pragma unroll
    for (int kk = 0; kk < INNER; kk += 32) {
        v16h a = load_a_frag(AO, INNER, mt + lm, kk, hg);
        v16h b = load_b_frag(brow, kk, hg);
        acc = wmma_f16(a, b, acc);
    }
    const float bias = bout[nt + lm];
#pragma unroll
    for (int r = 0; r < 8; ++r)
        out[(size_t)(mt + r + 8 * hg) * QDIM + nt + lm] = acc[r] + bias;
}

// ---------------------------------------------------------------------------
extern "C" void kernel_launch(void* const* d_in, const int* in_sizes, int n_in,
                              void* d_out, int out_size, void* d_ws, size_t ws_size,
                              hipStream_t stream) {
    const float* x    = (const float*)d_in[0];   // [1,4096,320]
    const float* gm   = (const float*)d_in[1];   // [1,3,64,64]
    const float* Wq   = (const float*)d_in[2];   // [320,512]
    const float* Wk   = (const float*)d_in[3];
    const float* Wv   = (const float*)d_in[4];
    const float* Wout = (const float*)d_in[5];   // [512,320]
    const float* bout = (const float*)d_in[6];   // [320]
    float* out = (float*)d_out;                  // [4096,320]

    // workspace carve-up (256B-aligned chunks)
    char* ws = (char*)d_ws;
    size_t off = 0;
    auto take = [&](size_t bytes) { char* p = ws + off; off += (bytes + 255) & ~(size_t)255; return p; };
    half_t* xh   = (half_t*)take((size_t)N_TOK * QDIM * 2);
    half_t* WqT  = (half_t*)take((size_t)INNER * QDIM * 2);
    half_t* WkT  = (half_t*)take((size_t)INNER * QDIM * 2);
    half_t* WvT  = (half_t*)take((size_t)INNER * QDIM * 2);
    half_t* WoT  = (half_t*)take((size_t)QDIM * INNER * 2);
    half_t* Qg   = (half_t*)take((size_t)NHEADS * N_TOK * DH * 2);
    half_t* Kg   = (half_t*)take((size_t)NHEADS * N_TOK * DH * 2);
    half_t* VTg  = (half_t*)take((size_t)NHEADS * DH * N_TOK * 2);
    half_t* AO   = (half_t*)take((size_t)N_TOK * INNER * 2);
    unsigned int* codes = (unsigned int*)take((size_t)N_TOK * 4);
    (void)ws_size; (void)in_sizes; (void)n_in; (void)out_size;

    // prep
    {
        int n = N_TOK * QDIM;
        k_cvt_f16<<<(n + 255) / 256, 256, 0, stream>>>(x, xh, n);
    }
    {
        int n = QDIM * INNER;
        k_transpose_f16<<<(n + 255) / 256, 256, 0, stream>>>(Wq, WqT, QDIM, INNER);
        k_transpose_f16<<<(n + 255) / 256, 256, 0, stream>>>(Wk, WkT, QDIM, INNER);
        k_transpose_f16<<<(n + 255) / 256, 256, 0, stream>>>(Wv, WvT, QDIM, INNER);
        k_transpose_f16<<<(n + 255) / 256, 256, 0, stream>>>(Wout, WoT, INNER, QDIM);
    }
    k_codes<<<(N_TOK + 255) / 256, 256, 0, stream>>>(gm, codes);

    // QKV projection
    k_qkv<<<dim3(N_TOK / 16, INNER / 64, 3), 128, 0, stream>>>(xh, WqT, WkT, WvT, Qg, Kg, VTg);

    // flash attention
    k_attn<<<dim3(N_TOK / 64, NHEADS), 128, 0, stream>>>(Qg, Kg, VTg, codes, AO);

    // output projection
    k_proj<<<dim3(N_TOK / 16, QDIM / 64), 128, 0, stream>>>(AO, WoT, bout, out);
}
